// SwitchableQuantizedGPT2_90228672954736
// MI455X (gfx1250) — matrix operand order repaired and checked by reading, not compiled, not run on hardware
//
#include <hip/hip_runtime.h>
#include <cstdint>

// ---------------- model constants ----------------
#define Lnum 8
#define Cdim 1024
#define Hn   16
#define Dh   64
#define Vv   50257
#define Bb   2
#define Tt   1024
#define Mrows (Bb*Tt)     // 2048
#define Rr   16
#define SCALING 2.0f      // 32 / 16

typedef __attribute__((ext_vector_type(16))) _Float16 v16h;
typedef __attribute__((ext_vector_type(8)))  float    v8f;
typedef __attribute__((ext_vector_type(4)))  unsigned u32x4;
typedef __attribute__((ext_vector_type(8)))  int      i32x8;
typedef __attribute__((ext_vector_type(4)))  int      i32x4;

union Frag { unsigned u[8]; v16h v; };

// ---- float <-> order-preserving uint encoding (for min/max atomics) ----
__device__ inline unsigned encf(float f){
  unsigned u = __float_as_uint(f);
  return (u >> 31) ? ~u : (u | 0x80000000u);
}
__device__ inline float decf(unsigned e){
  unsigned u = (e >> 31) ? (e & 0x7FFFFFFFu) : ~e;
  return __uint_as_float(u);
}

// ---- fake-quant helpers (reciprocal-multiply form) ----
__device__ inline float fq_act(float v, float s, float rs, float zp){
  float q = rintf(v * rs) + zp;
  q = fminf(fmaxf(q, 0.f), 255.f);
  return (q - zp) * s;
}
__device__ inline float fq_wt(float v, float s, float rs){
  float q = rintf(v * rs);
  q = fminf(fmaxf(q, -127.f), 127.f);
  return q * s;
}
__device__ inline unsigned packh(float a, float b){
  union { _Float16 h[2]; unsigned u; } p;
  p.h[0] = (_Float16)a; p.h[1] = (_Float16)b;
  return p.u;
}

// ---- TDM: async-load a [tile_h=128 x tile_w=32] fp32 tile into LDS ----
// remW/remH are remaining tensor extents from the tile origin (OOB reads -> 0).
__device__ inline void tdm_load_tile(const float* gsrc, unsigned lds_off,
                                     int remW, int remH, int ld)
{
  unsigned long long ga = (unsigned long long)gsrc;
  unsigned w = remW > 0 ? (unsigned)remW : 0u;
  unsigned h = remH > 0 ? (unsigned)remH : 0u;
  u32x4 g0;
  g0[0] = 1u;                                   // count=1 (valid user descriptor)
  g0[1] = lds_off;                              // lds_addr (bytes)
  g0[2] = (unsigned)ga;                         // global_addr[31:0]
  g0[3] = (unsigned)((ga >> 32) & 0x01FFFFFFull) | (2u << 30); // addr[56:32] | type=2
  i32x8 g1;
  g1[0] = (int)(2u << 16);                      // workgroup_mask=0, data_size=2 (4B)
  g1[1] = (int)((w & 0xFFFFu) << 16);           // tensor_dim0 low16 @ bits 63:48
  g1[2] = (int)(((w >> 16) & 0xFFFFu) | ((h & 0xFFFFu) << 16)); // dim0 hi | dim1 lo
  g1[3] = (int)(((h >> 16) & 0xFFFFu) | (32u << 16));           // dim1 hi | tile_dim0=32
  g1[4] = (int)128u;                            // tile_dim1=128, tile_dim2=0
  g1[5] = ld;                                   // tensor_dim0_stride (elements), low32
  g1[6] = 0;                                    // stride hi16 = 0, dim1_stride lo = 0
  g1[7] = 0;
  i32x4 z4 = {0,0,0,0};
#if defined(__clang_major__) && (__clang_major__ >= 23)
  i32x8 z8 = {0,0,0,0,0,0,0,0};
  __builtin_amdgcn_tensor_load_to_lds(g0, g1, z4, z4, z8, 0);
#else
  __builtin_amdgcn_tensor_load_to_lds(g0, g1, z4, z4, 0);
#endif
}

// ---------------- scalar slots init ----------------
__global__ void k_init(unsigned* slots){
  int i = threadIdx.x;
  if (i < 8) slots[i] = (i & 1) ? 0xFFFFFFFFu : 0u;
}

// ---------------- min/max over a column range ----------------
__global__ __launch_bounds__(256) void k_minmax(const float* __restrict__ base,
                                                int rows, int c0, int c1, int ld,
                                                unsigned* slots, int idx){
  const int W = c1 - c0;
  float mx = -3.4e38f, mn = 3.4e38f;
  for (int r = blockIdx.x; r < rows; r += gridDim.x){
    const float* p = base + (size_t)r * ld + c0;
    for (int c = threadIdx.x; c < W; c += 256){
      float v = p[c];
      mx = fmaxf(mx, v); mn = fminf(mn, v);
    }
  }
  __shared__ float smx[256], smn[256];
  smx[threadIdx.x] = mx; smn[threadIdx.x] = mn; __syncthreads();
  for (int k = 128; k > 0; k >>= 1){
    if (threadIdx.x < (unsigned)k){
      smx[threadIdx.x] = fmaxf(smx[threadIdx.x], smx[threadIdx.x + k]);
      smn[threadIdx.x] = fminf(smn[threadIdx.x], smn[threadIdx.x + k]);
    }
    __syncthreads();
  }
  if (threadIdx.x == 0){
    atomicMax(&slots[idx],     encf(smx[0]));
    atomicMin(&slots[idx + 1], encf(smn[0]));
  }
}

// ---------------- absmax over a flat weight tensor ----------------
__global__ __launch_bounds__(256) void k_absmax(const float* __restrict__ p,
                                                long long n, unsigned* slots, int idx){
  float mx = 0.f;
  for (long long i = (long long)blockIdx.x * blockDim.x + threadIdx.x; i < n;
       i += (long long)gridDim.x * blockDim.x)
    mx = fmaxf(mx, fabsf(p[i]));
  __shared__ float smx[256];
  smx[threadIdx.x] = mx; __syncthreads();
  for (int k = 128; k > 0; k >>= 1){
    if (threadIdx.x < (unsigned)k)
      smx[threadIdx.x] = fmaxf(smx[threadIdx.x], smx[threadIdx.x + k]);
    __syncthreads();
  }
  if (threadIdx.x == 0) atomicMax(&slots[idx], __float_as_uint(smx[0]));
}

// ---------------- embedding ----------------
__global__ __launch_bounds__(256) void k_embed(const int* __restrict__ ids,
                                               const float* __restrict__ wte,
                                               const float* __restrict__ wpe,
                                               float* __restrict__ h){
  int m = blockIdx.x;
  int t = m & (Tt - 1);
  int id = ids[m];
  const float* we = wte + (size_t)id * Cdim;
  const float* pe = wpe + (size_t)t * Cdim;
  for (int i = threadIdx.x; i < Cdim; i += 256)
    h[(size_t)m * Cdim + i] = we[i] + pe[i];
}

// ---------------- layernorm (one row per block) ----------------
__global__ __launch_bounds__(256) void k_ln(const float* __restrict__ in,
                                            const float* __restrict__ w,
                                            const float* __restrict__ b,
                                            float* __restrict__ out, int C){
  __shared__ float red[256];
  size_t row = blockIdx.x;
  const float* xr = in + row * C;
  float s = 0.f, ss = 0.f;
  for (int i = threadIdx.x; i < C; i += 256){ float v = xr[i]; s += v; ss += v * v; }
  red[threadIdx.x] = s; __syncthreads();
  for (int k = 128; k > 0; k >>= 1){
    if (threadIdx.x < (unsigned)k) red[threadIdx.x] += red[threadIdx.x + k];
    __syncthreads();
  }
  float mean = red[0] / C;
  __syncthreads();
  red[threadIdx.x] = ss; __syncthreads();
  for (int k = 128; k > 0; k >>= 1){
    if (threadIdx.x < (unsigned)k) red[threadIdx.x] += red[threadIdx.x + k];
    __syncthreads();
  }
  float var = red[0] / C - mean * mean;
  float inv = rsqrtf(var + 1e-5f);
  for (int i = threadIdx.x; i < C; i += 256){
    float v = xr[i];
    out[row * C + i] = (v - mean) * inv * w[i] + b[i];
  }
}

// ---------------- exact GELU ----------------
__global__ __launch_bounds__(256) void k_gelu(float* __restrict__ p, long long n){
  long long i = (long long)blockIdx.x * blockDim.x + threadIdx.x;
  if (i < n){
    float v = p[i];
    p[i] = 0.5f * v * (1.f + erff(v * 0.70710678118654752f));
  }
}

// ---------------- fused causal attention (online softmax, fp32) ----------------
__global__ __launch_bounds__(256) void k_attn(const float* __restrict__ qkv,
                                              float* __restrict__ o,
                                              const unsigned* __restrict__ slots){
  int gid = blockIdx.x * blockDim.x + threadIdx.x;
  int q  = gid & (Tt - 1);
  int hh = (gid >> 10) & (Hn - 1);
  int b  = gid >> 14;
  float kmx = decf(slots[2]), kmn = decf(slots[3]);
  float ks  = fmaxf((kmx - kmn) * (1.f / 255.f), 1e-8f);
  float rks = 1.f / ks;
  float kz  = rintf(-kmn * rks);
  float vmx = decf(slots[4]), vmn = decf(slots[5]);
  float vs  = fmaxf((vmx - vmn) * (1.f / 255.f), 1e-8f);
  float rvs = 1.f / vs;
  float vz  = rintf(-vmn * rvs);

  const float* qp = qkv + ((size_t)(b * Tt + q)) * (3 * Cdim) + hh * Dh;
  float qreg[Dh];
#pragma unroll
  for (int d = 0; d < Dh; d++) qreg[d] = qp[d];
  float m = -3.0e38f, s = 0.f;
  float acc[Dh];
#pragma unroll
  for (int d = 0; d < Dh; d++) acc[d] = 0.f;

  for (int t = 0; t <= q; t++){
    const float* kp = qkv + ((size_t)(b * Tt + t)) * (3 * Cdim) + Cdim + hh * Dh;
    const float* vp = kp + Cdim;
    float dot = 0.f;
#pragma unroll
    for (int d = 0; d < Dh; d++) dot += qreg[d] * fq_act(kp[d], ks, rks, kz);
    dot *= 0.125f;
    float mn2  = fmaxf(m, dot);
    float corr = __expf(m - mn2);
    float p    = __expf(dot - mn2);
    s = s * corr + p;
#pragma unroll
    for (int d = 0; d < Dh; d++) acc[d] = acc[d] * corr + p * fq_act(vp[d], vs, rvs, vz);
    m = mn2;
  }
  float inv = 1.f / s;
  float* op = o + ((size_t)(b * Tt + q)) * Cdim + hh * Dh;
#pragma unroll
  for (int d = 0; d < Dh; d++) op[d] = acc[d] * inv;
}

// -------- WMMA GEMM with TDM-streamed B tiles ----------------------------------
// Out = alpha * fq(A)[MxK] * fq(B)[NxK]^T + bias + res
// Block tile 64x128, 8 waves, 4 WMMA tiles/wave. B (weights) streamed by the
// Tensor Data Mover into a double-buffered fp32 LDS tile; quant+f16 conversion
// happens at fragment build. A (activations) staged via VALU fq -> f16 LDS.
template<bool AL>
__global__ __launch_bounds__(256) void k_gemm(
    const float* __restrict__ A, int lda,
    const float* __restrict__ Bm, int ldb,
    float* __restrict__ Out, int ldo,
    const float* __restrict__ bias,
    const float* __restrict__ res,
    int M, int N, int K, float alpha,
    const unsigned* __restrict__ slots, int aq, int wq)
{
  __shared__ unsigned As[64][17];        // 64 x 32 f16 pair-packed (+pad)
  __shared__ float    Bs32[2][128 * 32]; // double-buffered TDM destination

  const int tid  = threadIdx.x;
  const int lane = tid & 31, wave = tid >> 5;
  const int m0 = blockIdx.x * 64, n0 = blockIdx.y * 128;

  float as = 1.f, ras = 1.f, az = 0.f, wsc = 1.f, rwsc = 1.f;
  if (aq >= 0){
    float mx = decf(slots[aq]), mn = decf(slots[aq + 1]);
    as  = fmaxf((mx - mn) * (1.f / 255.f), 1e-8f);
    ras = 1.f / as;
    az  = rintf(-mn * ras);
  }
  if (wq >= 0){
    wsc  = fmaxf(__uint_as_float(slots[wq]) * (1.f / 127.f), 1e-8f);
    rwsc = 1.f / wsc;
  }

  const int arow = tid >> 2, ac0 = (tid & 3) * 8;
  const int tm = wave & 3, tnb = (wave >> 2) * 4;
  const int half = lane >> 4, li = lane & 15;

  const unsigned ldsB0 = (unsigned)(unsigned long long)(&Bs32[0][0]);
  const unsigned bufBytes = 128u * 32u * 4u;

  v8f acc0 = {}, acc1 = {}, acc2 = {}, acc3 = {};

  // prologue: DMA first B tile into buffer 0
  if (wave == 0)
    tdm_load_tile(Bm + (size_t)n0 * ldb, ldsB0, K, N - n0, ldb);

  int cur = 0;
  for (int k0 = 0; k0 < K; k0 += 32){
    const bool hasNext = (k0 + 32) < K;
    if (wave == 0 && hasNext)
      tdm_load_tile(Bm + (size_t)n0 * ldb + (k0 + 32),
                    ldsB0 + (unsigned)(cur ^ 1) * bufBytes,
                    K - (k0 + 32), N - n0, ldb);

    // ---- stage A tile (64x32): global load -> fq -> f16 pairs in LDS ----
    {
      int gm = m0 + arow;
      const float* src = A + (size_t)gm * lda + k0 + ac0;
      float4 f0 = {0.f,0.f,0.f,0.f}, f1 = {0.f,0.f,0.f,0.f};
      if (AL){
        f0 = *(const float4*)src;
        f1 = *(const float4*)(src + 4);
        if (hasNext) __builtin_prefetch(src + 32);
      } else {
        bool ok = gm < M;
        if (ok && k0 + ac0 + 4 <= K) f0 = *(const float4*)src;
        if (ok && k0 + ac0 + 8 <= K) f1 = *(const float4*)(src + 4);
      }
      float v[8] = {f0.x,f0.y,f0.z,f0.w,f1.x,f1.y,f1.z,f1.w};
      if (aq >= 0){
#pragma unroll
        for (int j = 0; j < 8; j++) v[j] = fq_act(v[j], as, ras, az);
      }
#pragma unroll
      for (int j = 0; j < 4; j++) As[arow][(ac0 >> 1) + j] = packh(v[2*j], v[2*j+1]);
    }

    // gate current B buffer: wave0 waits TENSORcnt, block syncs
    if (wave == 0){
      if (hasNext) __builtin_amdgcn_s_wait_tensorcnt(1);
      else         __builtin_amdgcn_s_wait_tensorcnt(0);
    }
    __syncthreads();

    // ---- A fragment (f16 pairs, contiguous k = 16*half + s, matching B) ----
    Frag fa;
#pragma unroll
    for (int j = 0; j < 8; j++){
      int pa = half * 8 + j;
      fa.u[j] = As[tm * 16 + li][pa];
    }
    // ---- B fragments: fp32 pairs from TDM buffer -> fq_wt -> f16 pack ----
    Frag fb0, fb1, fb2, fb3;
    const float* Bc = &Bs32[cur][0];
    const int r0 = ((tnb + 0) * 16 + li) * 32;
    const int r1 = ((tnb + 1) * 16 + li) * 32;
    const int r2 = ((tnb + 2) * 16 + li) * 32;
    const int r3 = ((tnb + 3) * 16 + li) * 32;
    if (wq >= 0){
#pragma unroll
      for (int j = 0; j < 8; j++){
        int col = half * 16 + 2 * j;
        float2 p0 = *(const float2*)&Bc[r0 + col];
        float2 p1 = *(const float2*)&Bc[r1 + col];
        float2 p2 = *(const float2*)&Bc[r2 + col];
        float2 p3 = *(const float2*)&Bc[r3 + col];
        fb0.u[j] = packh(fq_wt(p0.x, wsc, rwsc), fq_wt(p0.y, wsc, rwsc));
        fb1.u[j] = packh(fq_wt(p1.x, wsc, rwsc), fq_wt(p1.y, wsc, rwsc));
        fb2.u[j] = packh(fq_wt(p2.x, wsc, rwsc), fq_wt(p2.y, wsc, rwsc));
        fb3.u[j] = packh(fq_wt(p3.x, wsc, rwsc), fq_wt(p3.y, wsc, rwsc));
      }
    } else {
#pragma unroll
      for (int j = 0; j < 8; j++){
        int col = half * 16 + 2 * j;
        float2 p0 = *(const float2*)&Bc[r0 + col];
        float2 p1 = *(const float2*)&Bc[r1 + col];
        float2 p2 = *(const float2*)&Bc[r2 + col];
        float2 p3 = *(const float2*)&Bc[r3 + col];
        fb0.u[j] = packh(p0.x, p0.y);
        fb1.u[j] = packh(p1.x, p1.y);
        fb2.u[j] = packh(p2.x, p2.y);
        fb3.u[j] = packh(p3.x, p3.y);
      }
    }

    acc0 = __builtin_amdgcn_wmma_f32_16x16x32_f16(false, fa.v, false, fb0.v, (short)0, acc0, false, false);
    acc1 = __builtin_amdgcn_wmma_f32_16x16x32_f16(false, fa.v, false, fb1.v, (short)0, acc1, false, false);
    acc2 = __builtin_amdgcn_wmma_f32_16x16x32_f16(false, fa.v, false, fb2.v, (short)0, acc2, false, false);
    acc3 = __builtin_amdgcn_wmma_f32_16x16x32_f16(false, fa.v, false, fb3.v, (short)0, acc3, false, false);
    __syncthreads();
    cur ^= 1;
  }

  // ---- epilogue: C/D layout VGPR r -> row half*8+r, col = li ----
#pragma unroll
  for (int r = 0; r < 8; r++){
    int gm = m0 + tm * 16 + half * 8 + r;
    if (AL || gm < M){
      size_t ro = (size_t)gm * ldo;
      int gnb = n0 + tnb * 16 + li;
      {
        int gn = gnb;
        if (AL || gn < N){
          float v = alpha * acc0[r];
          if (bias) v += bias[gn];
          if (res)  v += res[ro + gn];
          Out[ro + gn] = v;
        }
      }
      {
        int gn = gnb + 16;
        if (AL || gn < N){
          float v = alpha * acc1[r];
          if (bias) v += bias[gn];
          if (res)  v += res[ro + gn];
          Out[ro + gn] = v;
        }
      }
      {
        int gn = gnb + 32;
        if (AL || gn < N){
          float v = alpha * acc2[r];
          if (bias) v += bias[gn];
          if (res)  v += res[ro + gn];
          Out[ro + gn] = v;
        }
      }
      {
        int gn = gnb + 48;
        if (AL || gn < N){
          float v = alpha * acc3[r];
          if (bias) v += bias[gn];
          if (res)  v += res[ro + gn];
          Out[ro + gn] = v;
        }
      }
    }
  }
}

// ---------------- host orchestration ----------------
extern "C" void kernel_launch(void* const* d_in, const int* in_sizes, int n_in,
                              void* d_out, int out_size, void* d_ws, size_t ws_size,
                              hipStream_t stream)
{
  (void)in_sizes; (void)n_in; (void)out_size; (void)ws_size;

  const int*   ids     = (const int*)  d_in[0];
  const float* wte     = (const float*)d_in[1];
  const float* wpe     = (const float*)d_in[2];
  const float* ln1_w   = (const float*)d_in[3];
  const float* ln1_b   = (const float*)d_in[4];
  const float* attn_W  = (const float*)d_in[5];
  const float* attn_b  = (const float*)d_in[6];
  const float* attn_A  = (const float*)d_in[7];
  const float* attn_Bm = (const float*)d_in[8];
  const float* aproj_W = (const float*)d_in[9];
  const float* aproj_b = (const float*)d_in[10];
  const float* aproj_A = (const float*)d_in[11];
  const float* aproj_Bm= (const float*)d_in[12];
  const float* ln2_w   = (const float*)d_in[13];
  const float* ln2_b   = (const float*)d_in[14];
  const float* fc_W    = (const float*)d_in[15];
  const float* fc_b    = (const float*)d_in[16];
  const float* fc_A    = (const float*)d_in[17];
  const float* fc_Bm   = (const float*)d_in[18];
  const float* mp_W    = (const float*)d_in[19];
  const float* mp_b    = (const float*)d_in[20];
  const float* mp_A    = (const float*)d_in[21];
  const float* mp_Bm   = (const float*)d_in[22];
  const float* lnf_w   = (const float*)d_in[23];
  const float* lnf_b   = (const float*)d_in[24];

  // workspace layout (~49 MB)
  char* wsp = (char*)d_ws;
  unsigned* slots = (unsigned*)wsp;                 wsp += 1024;
  float* h  = (float*)wsp;  wsp += (size_t)Mrows * Cdim     * 4;
  float* x  = (float*)wsp;  wsp += (size_t)Mrows * Cdim     * 4;
  float* y  = (float*)wsp;  wsp += (size_t)Mrows * 4 * Cdim * 4;
  float* t1 = (float*)wsp;  wsp += (size_t)Mrows * Rr       * 4;

  auto gemm = [&](const float* Ap, int lda, const float* Bp, int ldb, float* Op, int ldo,
                  const float* bias, const float* res, int M, int N, int K,
                  float alpha, int aq, int wq){
    dim3 g((M + 63) / 64, (N + 127) / 128);
    bool al = (M % 64 == 0) && (N % 128 == 0) && (K % 32 == 0);
    if (al)
      k_gemm<true><<<g, dim3(256), 0, stream>>>(Ap, lda, Bp, ldb, Op, ldo, bias, res,
                                                M, N, K, alpha, slots, aq, wq);
    else
      k_gemm<false><<<g, dim3(256), 0, stream>>>(Ap, lda, Bp, ldb, Op, ldo, bias, res,
                                                 M, N, K, alpha, slots, aq, wq);
  };
  auto init = [&](){ k_init<<<dim3(1), dim3(32), 0, stream>>>(slots); };
  auto mm = [&](const float* p, int rows, int cc0, int cc1, int ld, int idx){
    k_minmax<<<dim3(512), dim3(256), 0, stream>>>(p, rows, cc0, cc1, ld, slots, idx);
  };
  auto am = [&](const float* p, long long n, int idx){
    k_absmax<<<dim3(512), dim3(256), 0, stream>>>(p, n, slots, idx);
  };

  k_embed<<<dim3(Mrows), dim3(256), 0, stream>>>(ids, wte, wpe, h);

  for (int l = 0; l < Lnum; l++){
    const float* l1w = ln1_w   + (size_t)l * Cdim;
    const float* l1b = ln1_b   + (size_t)l * Cdim;
    const float* aW  = attn_W  + (size_t)l * 3 * Cdim * Cdim;
    const float* ab  = attn_b  + (size_t)l * 3 * Cdim;
    const float* aA  = attn_A  + (size_t)l * Rr * Cdim;
    const float* aB  = attn_Bm + (size_t)l * 3 * Cdim * Rr;
    const float* pW  = aproj_W + (size_t)l * Cdim * Cdim;
    const float* pb  = aproj_b + (size_t)l * Cdim;
    const float* pA  = aproj_A + (size_t)l * Rr * Cdim;
    const float* pB  = aproj_Bm+ (size_t)l * Cdim * Rr;
    const float* l2w = ln2_w   + (size_t)l * Cdim;
    const float* l2b = ln2_b   + (size_t)l * Cdim;
    const float* fW  = fc_W    + (size_t)l * 4 * Cdim * Cdim;
    const float* fb  = fc_b    + (size_t)l * 4 * Cdim;
    const float* fA  = fc_A    + (size_t)l * Rr * Cdim;
    const float* fB  = fc_Bm   + (size_t)l * 4 * Cdim * Rr;
    const float* mW  = mp_W    + (size_t)l * Cdim * 4 * Cdim;
    const float* mb  = mp_b    + (size_t)l * Cdim;
    const float* mA  = mp_A    + (size_t)l * Rr * 4 * Cdim;
    const float* mB  = mp_Bm   + (size_t)l * Cdim * Rr;

    // ---- attention sublayer ----
    k_ln<<<dim3(Mrows), dim3(256), 0, stream>>>(h, l1w, l1b, x, Cdim);
    init(); mm(x, Mrows, 0, Cdim, Cdim, 0); am(aW, (long long)3 * Cdim * Cdim, 6);
    gemm(x, Cdim, aW, Cdim, y, 3 * Cdim, ab, nullptr, Mrows, 3 * Cdim, Cdim, 1.f, 0, 6);
    gemm(x, Cdim, aA, Cdim, t1, Rr, nullptr, nullptr, Mrows, Rr, Cdim, 1.f, 0, -1);
    gemm(t1, Rr, aB, Rr, y, 3 * Cdim, nullptr, y, Mrows, 3 * Cdim, Rr, SCALING, -1, -1);
    init();
    mm(y, Mrows,     Cdim, 2 * Cdim, 3 * Cdim, 2);   // k min/max
    mm(y, Mrows, 2 * Cdim, 3 * Cdim, 3 * Cdim, 4);   // v min/max
    k_attn<<<dim3(Mrows * Hn / 256), dim3(256), 0, stream>>>(y, x, slots);
    init(); mm(x, Mrows, 0, Cdim, Cdim, 0); am(pW, (long long)Cdim * Cdim, 6);
    gemm(x, Cdim, pW, Cdim, h, Cdim, pb, h, Mrows, Cdim, Cdim, 1.f, 0, 6);
    gemm(x, Cdim, pA, Cdim, t1, Rr, nullptr, nullptr, Mrows, Rr, Cdim, 1.f, 0, -1);
    gemm(t1, Rr, pB, Rr, h, Cdim, nullptr, h, Mrows, Cdim, Rr, SCALING, -1, -1);

    // ---- MLP sublayer ----
    k_ln<<<dim3(Mrows), dim3(256), 0, stream>>>(h, l2w, l2b, x, Cdim);
    init(); mm(x, Mrows, 0, Cdim, Cdim, 0); am(fW, (long long)4 * Cdim * Cdim, 6);
    gemm(x, Cdim, fW, Cdim, y, 4 * Cdim, fb, nullptr, Mrows, 4 * Cdim, Cdim, 1.f, 0, 6);
    gemm(x, Cdim, fA, Cdim, t1, Rr, nullptr, nullptr, Mrows, Rr, Cdim, 1.f, 0, -1);
    gemm(t1, Rr, fB, Rr, y, 4 * Cdim, nullptr, y, Mrows, 4 * Cdim, Rr, SCALING, -1, -1);
    k_gelu<<<dim3((Mrows * 4 * Cdim + 255) / 256), dim3(256), 0, stream>>>(
        y, (long long)Mrows * 4 * Cdim);
    init(); mm(y, Mrows, 0, 4 * Cdim, 4 * Cdim, 0); am(mW, (long long)Cdim * 4 * Cdim, 6);
    gemm(y, 4 * Cdim, mW, 4 * Cdim, h, Cdim, mb, h, Mrows, Cdim, 4 * Cdim, 1.f, 0, 6);
    gemm(y, 4 * Cdim, mA, 4 * Cdim, t1, Rr, nullptr, nullptr, Mrows, Rr, 4 * Cdim, 1.f, 0, -1);
    gemm(t1, Rr, mB, Rr, h, Cdim, nullptr, h, Mrows, Cdim, Rr, SCALING, -1, -1);
  }

  // ---- final LN + tied lm_head ----
  k_ln<<<dim3(Mrows), dim3(256), 0, stream>>>(h, lnf_w, lnf_b, x, Cdim);
  gemm(x, Cdim, wte, Cdim, (float*)d_out, Vv, nullptr, nullptr,
       Mrows, Vv, Cdim, 1.f, -1, -1);
}
